// SparseFFN_22351009808935
// MI455X (gfx1250) — matrix operand order repaired and checked by reading, not compiled, not run on hardware
//
#include <hip/hip_runtime.h>
#include <hip/hip_bf16.h>
#include <math.h>

// ---------------------------------------------------------------------------
// SparseFFN for MI455X (gfx1250):  out = GELU(x @ (w_up*m_up)^T) @ (w_down*m_down)^T
// Masks + fp32->bf16 fused into one prep pass, then two bf16 WMMA GEMMs
// (v_wmma_f32_16x16x32_bf16, fp32 accum). GELU fused into GEMM1 epilogue,
// intermediate h kept bf16 in workspace.
//
// GEMM: 64x64 tile per wave = 4 A-frags x 4 B-frags -> 16 WMMAs per K=32 step
// against 16 b128 loads (32 FLOP/byte of cache traffic). Grouped tile
// rasterization keeps concurrent blocks sharing A/B slabs in L2.
// ---------------------------------------------------------------------------

typedef __attribute__((ext_vector_type(16))) __bf16 v16bf;
typedef __attribute__((ext_vector_type(8)))  __bf16 v8bf;
typedef __attribute__((ext_vector_type(8)))  float  v8f;

static constexpr int D_MODEL = 2048;
static constexpr int D_FF    = 8192;
static constexpr int TOKENS  = 8192;

// ---------------------------------------------------------------------------
// Prep kernels: fp32 -> bf16 convert, optionally applying a binary mask.
// ---------------------------------------------------------------------------
__global__ void k_cvt_bf16(const float* __restrict__ src,
                           __bf16* __restrict__ dst, int n) {
  int i = (blockIdx.x * blockDim.x + threadIdx.x) * 8;
  if (i >= n) return;
#pragma unroll
  for (int j = 0; j < 8; ++j) dst[i + j] = (__bf16)src[i + j];
}

__global__ void k_cvt_mask_bf16(const float* __restrict__ src,
                                const unsigned char* __restrict__ mask,
                                __bf16* __restrict__ dst, int n) {
  int i = (blockIdx.x * blockDim.x + threadIdx.x) * 8;
  if (i >= n) return;
#pragma unroll
  for (int j = 0; j < 8; ++j) {
    float v = src[i + j];
    dst[i + j] = mask[i + j] ? (__bf16)v : (__bf16)0.0f;
  }
}

// ---------------------------------------------------------------------------
// Fragment loads (wave32), contiguous k order shared by A and B (not the ISA table order):
//   A 16x32 bf16: lane l -> row l%16; elems 0..7 = K 16*(l/16)+j,
//                                     elems 8..15 = K 16*(l/16)+8+j
//   B 32x16 bf16 (col n = W row):  lane l -> col l%16, K = 16*(l/16)..+15
//   C/D f32: vgpr r -> row r + 8*(l/16), col l%16
// ---------------------------------------------------------------------------
__device__ __forceinline__ v16bf load_a_frag(const __bf16* __restrict__ rowp,
                                             int k0, int half) {
  v8bf lo = *(const v8bf*)(rowp + k0 + 16 * half);
  v8bf hi = *(const v8bf*)(rowp + k0 + 8 + 16 * half);
  return __builtin_shufflevector(lo, hi, 0, 1, 2, 3, 4, 5, 6, 7,
                                 8, 9, 10, 11, 12, 13, 14, 15);
}

template <bool FUSE_GELU, typename OUT_T>
__device__ __forceinline__ void store_tile(OUT_T* __restrict__ C, size_t ldc,
                                           const v8f& acc, int row0, int col) {
#pragma unroll
  for (int r = 0; r < 8; ++r) {
    float v = acc[r];
    if (FUSE_GELU) v = 0.5f * v * (1.0f + erff(v * 0.70710678118654752f));
    C[(size_t)(row0 + r) * ldc + col] = (OUT_T)v;
  }
}

// C[M,N] = A[M,K] @ B[N,K]^T, 64x64 tile per wave, 8 waves/block.
template <bool FUSE_GELU, typename OUT_T>
__global__ __launch_bounds__(256) void k_gemm_bf16(
    const __bf16* __restrict__ A, const __bf16* __restrict__ B,
    OUT_T* __restrict__ C, int M, int N, int K) {
  const int lane = threadIdx.x & 31;
  const int half = lane >> 4;
  const int l16  = lane & 15;
  const int waveId = blockIdx.x * 8 + (threadIdx.x >> 5);

  // Grouped (L2-friendly) rasterization over 64x64 tiles.
  const int tilesM = M >> 6;
  const int tilesN = N >> 6;
  const int GROUP  = 8;
  const int groupSpan = GROUP * tilesN;
  const int groupId   = waveId / groupSpan;
  const int firstM    = groupId * GROUP;
  const int gsz       = (tilesM - firstM) < GROUP ? (tilesM - firstM) : GROUP;
  const int tmIdx     = firstM + (waveId % gsz);
  const int tnIdx     = (waveId % groupSpan) / gsz;
  const int tm = tmIdx * 64;
  const int tn = tnIdx * 64;
  if (tm >= M || tn >= N) return;

  const __bf16* aRow[4];
  const __bf16* bRow[4];
#pragma unroll
  for (int i = 0; i < 4; ++i) {
    aRow[i] = A + (size_t)(tm + 16 * i + l16) * K;
    bRow[i] = B + (size_t)(tn + 16 * i + l16) * K + half * 16;
  }

  v8f acc[4][4] = {};

  for (int k0 = 0; k0 < K; k0 += 32) {
    // Prefetch ahead (global_prefetch_b8): each lane covers a distinct row.
#pragma unroll
    for (int i = 0; i < 4; ++i) {
      __builtin_prefetch(aRow[i] + k0 + 1024, 0, 1);
      __builtin_prefetch(bRow[i] + k0 + 1024, 0, 1);
    }

    v16bf af[4], bf[4];
#pragma unroll
    for (int i = 0; i < 4; ++i) af[i] = load_a_frag(aRow[i], k0, half);
#pragma unroll
    for (int j = 0; j < 4; ++j) bf[j] = *(const v16bf*)(bRow[j] + k0);

#pragma unroll
    for (int i = 0; i < 4; ++i)
#pragma unroll
      for (int j = 0; j < 4; ++j)
        acc[i][j] = __builtin_amdgcn_wmma_f32_16x16x32_bf16(
            false, af[i], false, bf[j], (short)0, acc[i][j], false, false);
  }

  const int r0 = tm + 8 * half;
#pragma unroll
  for (int i = 0; i < 4; ++i)
#pragma unroll
    for (int j = 0; j < 4; ++j)
      store_tile<FUSE_GELU>(C, (size_t)N, acc[i][j], r0 + 16 * i,
                            tn + 16 * j + l16);
}

// ---------------------------------------------------------------------------
// Launch: prep (3 convert kernels) -> GEMM1+GELU (bf16 h) -> GEMM2 (fp32 out)
// Workspace: x_bf (32MiB) | wu_bf (32MiB) | wd_bf (32MiB) | h_bf (128MiB)
// ---------------------------------------------------------------------------
extern "C" void kernel_launch(void* const* d_in, const int* in_sizes, int n_in,
                              void* d_out, int out_size, void* d_ws, size_t ws_size,
                              hipStream_t stream) {
  (void)in_sizes; (void)n_in; (void)out_size; (void)ws_size;

  const float*         x      = (const float*)d_in[0];
  const float*         w_up   = (const float*)d_in[1];
  const float*         w_down = (const float*)d_in[2];
  const unsigned char* m_up   = (const unsigned char*)d_in[3];
  const unsigned char* m_down = (const unsigned char*)d_in[4];
  float*               out    = (float*)d_out;

  __bf16* xb  = (__bf16*)d_ws;
  __bf16* wub = xb  + (size_t)TOKENS * D_MODEL;
  __bf16* wdb = wub + (size_t)D_FF   * D_MODEL;
  __bf16* hb  = wdb + (size_t)D_MODEL * D_FF;

  // --- prep: fp32 -> bf16 (+mask fusion) ---
  {
    int n = TOKENS * D_MODEL;                       // 16M
    k_cvt_bf16<<<n / 8 / 256, 256, 0, stream>>>(x, xb, n);
  }
  {
    int n = D_FF * D_MODEL;                         // 16M
    k_cvt_mask_bf16<<<n / 8 / 256, 256, 0, stream>>>(w_up, m_up, wub, n);
  }
  {
    int n = D_MODEL * D_FF;                         // 16M
    k_cvt_mask_bf16<<<n / 8 / 256, 256, 0, stream>>>(w_down, m_down, wdb, n);
  }

  // --- GEMM1: h = GELU(x @ wu^T)  [8192 x 8192], K = 2048 ---
  {
    int waveTiles = (TOKENS / 64) * (D_FF / 64);    // 16384
    k_gemm_bf16<true, __bf16><<<waveTiles / 8, 256, 0, stream>>>(
        xb, wub, hb, TOKENS, D_FF, D_MODEL);
  }

  // --- GEMM2: out = h @ wd^T  [8192 x 2048], K = 8192 ---
  {
    int waveTiles = (TOKENS / 64) * (D_MODEL / 64); // 4096
    k_gemm_bf16<false, float><<<waveTiles / 8, 256, 0, stream>>>(
        hb, wdb, out, TOKENS, D_MODEL, D_FF);
  }
}
